// MOEAdapter_87256555585597
// MI455X (gfx1250) — matrix operand order repaired and checked by reading, not compiled, not run on hardware
//
#include <hip/hip_runtime.h>
#include <math.h>

// ---------------------------------------------------------------------------
// MoE adapter for MI455X (gfx1250, wave32).
// Dense-expert formulation (matches reference math): 77.3 GFLOP of GEMM,
// ~206 MB of HBM traffic -> compute bound; use bf16 WMMA w/ fp32 accumulate.
// ---------------------------------------------------------------------------

typedef __attribute__((ext_vector_type(8)))  float  v8f;
typedef __attribute__((ext_vector_type(4)))  __bf16 v4bf;
typedef __attribute__((ext_vector_type(8)))  __bf16 v8bf;
typedef __attribute__((ext_vector_type(16))) __bf16 v16bf;

#define C_DIM   768
#define E_DIM   4
#define HD_DIM  192
#define NTOK    32768        // B*H*W = 32*32*32
#define SA_STRIDE 776        // 768 + 8 halves pad (row = 1552 B, 16B aligned)
#define SH_STRIDE 200        // 192 + 8 halves pad (row = 400 B, 16B aligned)

static __device__ inline v8f wmma_bf16(v16bf a, v16bf b, v8f c) {
  // D = A(16x32 bf16) x B(32x16 bf16) + C(16x16 f32)
  return __builtin_amdgcn_wmma_f32_16x16x32_bf16(false, a, false, b,
                                                 (short)0, c, false, false);
}
static __device__ inline v16bf cat16(v8bf lo, v8bf hi) {
  return __builtin_shufflevector(lo, hi, 0,1,2,3,4,5,6,7,8,9,10,11,12,13,14,15);
}

// ---------------------------------------------------------------------------
// Kernel 1: router. One wave32 per token: logits -> softmax -> top2 -> renorm
// -> per-expert combine weight [NTOK, 4].
// ---------------------------------------------------------------------------
__global__ void moe_router_kernel(const float* __restrict__ x,
                                  const float* __restrict__ rw,
                                  const float* __restrict__ rb,
                                  float* __restrict__ combine) {
  const int wave = threadIdx.x >> 5;
  const int lane = threadIdx.x & 31;
  const int t = blockIdx.x * (blockDim.x >> 5) + wave;
  if (t >= NTOK) return;
  const float* xt = x + (size_t)t * C_DIM;
  float p0 = 0.f, p1 = 0.f, p2 = 0.f, p3 = 0.f;
  #pragma unroll 4
  for (int i = 0; i < C_DIM / 32; ++i) {
    const int c = lane + i * 32;
    const float xv = xt[c];
    p0 = fmaf(xv, rw[0 * C_DIM + c], p0);
    p1 = fmaf(xv, rw[1 * C_DIM + c], p1);
    p2 = fmaf(xv, rw[2 * C_DIM + c], p2);
    p3 = fmaf(xv, rw[3 * C_DIM + c], p3);
  }
  #pragma unroll
  for (int off = 16; off >= 1; off >>= 1) {
    p0 += __shfl_xor(p0, off, 32);
    p1 += __shfl_xor(p1, off, 32);
    p2 += __shfl_xor(p2, off, 32);
    p3 += __shfl_xor(p3, off, 32);
  }
  if (lane == 0) {
    float lg[4] = {p0 + rb[0], p1 + rb[1], p2 + rb[2], p3 + rb[3]};
    float mx = fmaxf(fmaxf(lg[0], lg[1]), fmaxf(lg[2], lg[3]));
    float pr[4]; float s = 0.f;
    #pragma unroll
    for (int e = 0; e < 4; ++e) { pr[e] = __expf(lg[e] - mx); s += pr[e]; }
    const float inv = 1.0f / s;
    #pragma unroll
    for (int e = 0; e < 4; ++e) pr[e] *= inv;
    int i0 = 0; float v0 = pr[0];
    #pragma unroll
    for (int e = 1; e < 4; ++e) if (pr[e] > v0) { v0 = pr[e]; i0 = e; }
    int i1 = -1; float v1 = -1.0f;
    #pragma unroll
    for (int e = 0; e < 4; ++e) if (e != i0 && pr[e] > v1) { v1 = pr[e]; i1 = e; }
    const float dn = 1.0f / (v0 + v1 + 1e-8f);
    float cb[4] = {0.f, 0.f, 0.f, 0.f};
    cb[i0] = v0 * dn; cb[i1] = v1 * dn;
    #pragma unroll
    for (int e = 0; e < 4; ++e) combine[(size_t)t * 4 + e] = cb[e];
  }
}

// ---------------------------------------------------------------------------
// Kernel 2: fc1_w / fc2_w fp32 -> bf16 (each 589,824 elements; L2-resident).
// ---------------------------------------------------------------------------
__global__ void moe_cvt_kernel(const float* __restrict__ f1w,
                               const float* __restrict__ f2w,
                               __bf16* __restrict__ f1wb,
                               __bf16* __restrict__ f2wb) {
  const int n = E_DIM * HD_DIM * C_DIM;
  for (int i = blockIdx.x * blockDim.x + threadIdx.x; i < n;
       i += gridDim.x * blockDim.x) {
    f1wb[i] = (__bf16)f1w[i];
    f2wb[i] = (__bf16)f2w[i];
  }
}

// ---------------------------------------------------------------------------
// Kernel 3: fused dense-MoE. One 4-wave block per 16-token tile.
//   per expert e: H = gelu(x @ fc1_e^T + b1) * combine[:,e]  (bf16 in LDS)
//                 out += H @ fc2_e^T                          (f32 in VGPRs)
//   out += sum_e combine[:,e]*fc2_b[e]  (folded into accumulator init)
// ---------------------------------------------------------------------------
__global__ __launch_bounds__(128, 1)
void moe_main_kernel(const float* __restrict__ x,
                     const __bf16* __restrict__ f1wb,
                     const float* __restrict__ f1b,
                     const __bf16* __restrict__ f2wb,
                     const float* __restrict__ f2b,
                     const float* __restrict__ combine,
                     float* __restrict__ out) {
  __shared__ __align__(16) __bf16 sA[16 * SA_STRIDE];   // x tile, bf16
  __shared__ __align__(16) __bf16 sH[16 * SH_STRIDE];   // hidden tile, bf16
  __shared__ float sComb[16 * 4];                       // combine weights
  __shared__ float sB1[E_DIM * HD_DIM];                 // fc1 bias
  __shared__ float sB2[E_DIM * C_DIM];                  // fc2 bias

  const int tid    = threadIdx.x;
  const int wave   = tid >> 5;
  const int lane   = tid & 31;
  const int laneHi = lane >> 4;      // 0: lanes 0-15, 1: lanes 16-31
  const int nloc   = lane & 15;      // N / column within 16-wide tile
  const int Ma     = lane & 15;      // A-fragment row (M)
  const int tileBase = blockIdx.x * 16;

  // ---- phase 0: stage x tile (fp32 -> bf16), combine, biases into LDS ----
  for (int idx = tid; idx < 16 * (C_DIM / 4); idx += 128) {
    const int row = idx / (C_DIM / 4);
    const int c4  = idx % (C_DIM / 4);
    const float4 v = *(const float4*)(x + (size_t)(tileBase + row) * C_DIM + c4 * 4);
    v4bf b;
    b[0] = (__bf16)v.x; b[1] = (__bf16)v.y; b[2] = (__bf16)v.z; b[3] = (__bf16)v.w;
    *(v4bf*)(&sA[row * SA_STRIDE + c4 * 4]) = b;
  }
  for (int idx = tid; idx < 64; idx += 128)   sComb[idx] = combine[(size_t)tileBase * 4 + idx];
  for (int idx = tid; idx < E_DIM * HD_DIM; idx += 128) sB1[idx] = f1b[idx];
  for (int idx = tid; idx < E_DIM * C_DIM; idx += 128)  sB2[idx] = f2b[idx];
  __syncthreads();

  // ---- init output accumulators with weighted fc2 bias term ----
  // each wave owns 12 of the 48 column tiles of out[16, 768]
  v8f acc2[12];
  #pragma unroll
  for (int t = 0; t < 12; ++t) {
    const int c = (wave * 12 + t) * 16 + nloc;
    #pragma unroll
    for (int r = 0; r < 8; ++r) {
      const int m = r + 8 * laneHi;
      float b = 0.f;
      #pragma unroll
      for (int e = 0; e < 4; ++e) b = fmaf(sComb[m * 4 + e], sB2[e * C_DIM + c], b);
      acc2[t][r] = b;
    }
  }

  // ---- expert loop ----
  for (int e = 0; e < E_DIM; ++e) {
    // GEMM1: H[16,192] = xA[16,768] @ fc1_e^T ; each wave owns 3 of 12 h-tiles
    v8f acc1[3];
    #pragma unroll
    for (int t = 0; t < 3; ++t)
      #pragma unroll
      for (int r = 0; r < 8; ++r) acc1[t][r] = 0.f;

    const __bf16* w1e = f1wb + (size_t)e * HD_DIM * C_DIM;
    for (int kk = 0; kk < C_DIM / 32; ++kk) {
      const __bf16* ap = &sA[Ma * SA_STRIDE + kk * 32 + laneHi * 8];
      const v16bf afr = cat16(*(const v8bf*)ap, *(const v8bf*)(ap + 16));
      #pragma unroll
      for (int t = 0; t < 3; ++t) {
        const int h = (wave * 3 + t) * 16 + nloc;   // B column = fc1 row (contiguous)
        const __bf16* bp = w1e + (size_t)h * C_DIM + kk * 32 + laneHi * 8;
        const v16bf bfr = cat16(*(const v8bf*)bp, *(const v8bf*)(bp + 16));
        acc1[t] = wmma_bf16(afr, bfr, acc1[t]);
      }
    }
    __syncthreads();   // previous expert's GEMM2 reads of sH are done

    // epilogue: bias + exact GELU + combine scale -> bf16 sH
    #pragma unroll
    for (int t = 0; t < 3; ++t) {
      const int col = (wave * 3 + t) * 16 + nloc;
      const float bias = sB1[e * HD_DIM + col];
      #pragma unroll
      for (int r = 0; r < 8; ++r) {
        const int m = r + 8 * laneHi;
        const float v = acc1[t][r] + bias;
        float g = 0.5f * v * (1.0f + erff(v * 0.70710678118654752f));
        g *= sComb[m * 4 + e];
        sH[m * SH_STRIDE + col] = (__bf16)g;
      }
    }
    __syncthreads();

    // GEMM2: out[16,768] += H[16,192] @ fc2_e^T
    const __bf16* w2e = f2wb + (size_t)e * C_DIM * HD_DIM;
    for (int kk = 0; kk < HD_DIM / 32; ++kk) {
      const __bf16* ap = &sH[Ma * SH_STRIDE + kk * 32 + laneHi * 8];
      const v16bf afr = cat16(*(const v8bf*)ap, *(const v8bf*)(ap + 16));
      #pragma unroll
      for (int t = 0; t < 12; ++t) {
        const int c = (wave * 12 + t) * 16 + nloc;  // B column = fc2 row (contiguous)
        const __bf16* bp = w2e + (size_t)c * HD_DIM + kk * 32 + laneHi * 8;
        const v16bf bfr = cat16(*(const v8bf*)bp, *(const v8bf*)(bp + 16));
        acc2[t] = wmma_bf16(afr, bfr, acc2[t]);
      }
    }
  }

  // ---- write out tile (coalesced per half-wave) ----
  #pragma unroll
  for (int t = 0; t < 12; ++t) {
    const int c = (wave * 12 + t) * 16 + nloc;
    #pragma unroll
    for (int r = 0; r < 8; ++r) {
      const int m = r + 8 * laneHi;
      out[(size_t)(tileBase + m) * C_DIM + c] = acc2[t][r];
    }
  }
}

// ---------------------------------------------------------------------------
extern "C" void kernel_launch(void* const* d_in, const int* in_sizes, int n_in,
                              void* d_out, int out_size, void* d_ws, size_t ws_size,
                              hipStream_t stream) {
  const float* x   = (const float*)d_in[0];
  const float* rw  = (const float*)d_in[1];
  const float* rb  = (const float*)d_in[2];
  const float* f1w = (const float*)d_in[3];
  const float* f1b = (const float*)d_in[4];
  const float* f2w = (const float*)d_in[5];
  const float* f2b = (const float*)d_in[6];
  float* out = (float*)d_out;

  char* ws = (char*)d_ws;
  float*  combine = (float*)ws;                               // 32768*4*4  = 524,288 B
  __bf16* f1wb    = (__bf16*)(ws + 524288);                   // 589824*2   = 1,179,648 B
  __bf16* f2wb    = (__bf16*)(ws + 524288 + 1179648);         // 589824*2   = 1,179,648 B

  // 1) router: 32768 tokens, one wave each (8 waves / 256-thread block)
  moe_router_kernel<<<NTOK / 8, 256, 0, stream>>>(x, rw, rb, combine);
  // 2) weight conversion to bf16 (exactly one element per thread)
  moe_cvt_kernel<<<1152, 512, 0, stream>>>(f1w, f2w, f1wb, f2wb);
  // 3) fused MoE: one block per 16-token tile
  moe_main_kernel<<<NTOK / 16, 128, 0, stream>>>(x, f1wb, f1b, f2wb, f2b, combine, out);
}
